// TraceModel_16716012716286
// MI455X (gfx1250) — compile-verified, matrix operand order checked by reading, not run on hardware
//
#include <hip/hip_runtime.h>

// ---------------------------------------------------------------------------
// Types for CDNA5 WMMA (wave32): v_wmma_f32_16x16x32_bf16  + TDM descriptors
// ---------------------------------------------------------------------------
typedef __bf16 bf16;
typedef __attribute__((ext_vector_type(16))) __bf16 v16bf;
typedef __attribute__((ext_vector_type(8)))  __bf16 bf16x8;
typedef __attribute__((ext_vector_type(4)))  __bf16 bf16x4;
typedef __attribute__((ext_vector_type(8)))  float  v8f;
typedef __attribute__((ext_vector_type(4)))  unsigned int u32x4;
typedef __attribute__((ext_vector_type(8)))  int          i32x8;
typedef __attribute__((ext_vector_type(4)))  int          i32x4;

#define N_NODES 100000
#define N_EDGES 1600000
#define EMB     128
#define OUTF    64
#define NEG_SLOPE 0.2f

// float -> bf16, round-to-nearest-even via bit trick
__device__ __forceinline__ bf16 f2bf(float f) {
  unsigned u = __float_as_uint(f);
  unsigned r = (u + 0x7FFFu + ((u >> 16) & 1u)) >> 16;
  unsigned short s = (unsigned short)r;
  bf16 b;
  __builtin_memcpy(&b, &s, 2);
  return b;
}

// order-preserving float<->uint key for atomic max over signed floats
__device__ __forceinline__ unsigned f2key(float f) {
  unsigned u = __float_as_uint(f);
  return (u & 0x80000000u) ? ~u : (u | 0x80000000u);
}
__device__ __forceinline__ float key2f(unsigned k) {
  return __uint_as_float((k & 0x80000000u) ? (k & 0x7FFFFFFFu) : ~k);
}

__device__ __forceinline__ void store_out(float* p, float v) { *p = v; }
__device__ __forceinline__ void store_out(bf16* p, float v)  { *p = f2bf(v); }

// Generic pointer to LDS byte offset: per ISA 10.2 aperture mapping the
// hardware uses addr[31:0] for LDS, so the low 32 bits are the LDS address.
__device__ __forceinline__ unsigned lds_off(const void* p) {
  return (unsigned)(unsigned long long)p;
}

// ---------------------------------------------------------------------------
// TDM: issue tensor_load_to_lds of a (rows x 32) bf16 tile of A[M,K] starting
// at gptr (tile start) into LDS offset ldsoff.  2D descriptor per ISA ch.8:
//   group0: count=1 | lds_addr | global_addr[56:0] | type=2
//   group1: data_size=1(2B) | tensor_dim0=K | tensor_dim1=M | tile=32 x rows
//           tensor_dim0_stride=K
// This toolchain (clang-23 / therock-10.0) exposes the 6-arg builtin:
//   (u32x4 g0, i32x8 g1, i32x4 g2, i32x4 g3, i32x8 pad, i32 cpol)
// ---------------------------------------------------------------------------
__device__ __forceinline__ void tdm_load_a(const bf16* gptr, unsigned ldsoff,
                                           int K, int M, int rows) {
  unsigned long long ga = (unsigned long long)gptr;
  u32x4 g0;
  g0[0] = 1u;                                   // count=1, user mode
  g0[1] = ldsoff;                               // lds_addr (bytes)
  g0[2] = (unsigned)ga;                         // global_addr[31:0]
  g0[3] = (unsigned)((ga >> 32) & 0x01FFFFFFu)  // global_addr[56:32]
          | (2u << 30);                         // type = 2 ("image")
  i32x8 g1;
  g1[0] = (int)(1u << 16);                      // data_size code 1 = 2 bytes
  g1[1] = (int)(((unsigned)K & 0xFFFFu) << 16); // tensor_dim0[15:0] @bit48
  g1[2] = (int)((((unsigned)K >> 16) & 0xFFFFu)        // tensor_dim0[31:16]
          | (((unsigned)M & 0xFFFFu) << 16));          // tensor_dim1[15:0]
  g1[3] = (int)((((unsigned)M >> 16) & 0xFFFFu)        // tensor_dim1[31:16]
          | (32u << 16));                              // tile_dim0 = 32
  g1[4] = rows;                                 // tile_dim1 = rows, tile_dim2=0
  g1[5] = K;                                    // tensor_dim0_stride[31:0]
  g1[6] = 0;                                    // stride hi / dim1_stride lo
  g1[7] = 0;
  i32x4 z4 = {0, 0, 0, 0};
  i32x8 z8 = {0, 0, 0, 0, 0, 0, 0, 0};
  __builtin_amdgcn_tensor_load_to_lds(g0, g1, z4, z4, z8, 0);
}

// ---------------------------------------------------------------------------
// Init: zero x3 accumulator + softmax denominators, max-keys = key(-inf)
// ---------------------------------------------------------------------------
__global__ __launch_bounds__(256) void gat_init(float* __restrict__ x3,
                                                float* __restrict__ denom,
                                                unsigned* __restrict__ maxkey,
                                                int n) {
  int idx = blockIdx.x * blockDim.x + threadIdx.x;
  if (idx < n * OUTF) x3[idx] = 0.0f;
  if (idx < n) { denom[idx] = 0.0f; maxkey[idx] = 0x007FFFFFu; /* key(-inf) */ }
}

// ---------------------------------------------------------------------------
// Weight convert+transpose: W[K,NOUT] f32 (row-major) -> Wt[NOUT,K] bf16
// ---------------------------------------------------------------------------
__global__ __launch_bounds__(256) void wt_convert(const float* __restrict__ W,
                                                  bf16* __restrict__ Wt,
                                                  int K, int NOUT) {
  int idx = blockIdx.x * blockDim.x + threadIdx.x;
  if (idx >= K * NOUT) return;
  int n = idx / K, k = idx % K;
  Wt[idx] = f2bf(W[(size_t)k * NOUT + n]);
}

// ---------------------------------------------------------------------------
// Node prep: x1/x2 rank-1 linears -> hcat[:,0:128] (bf16), embed -> bf16
// ---------------------------------------------------------------------------
__global__ __launch_bounds__(128) void node_prep(const float* __restrict__ x,
                                                 const float* __restrict__ w1,
                                                 const float* __restrict__ b1,
                                                 const float* __restrict__ w2,
                                                 const float* __restrict__ b2,
                                                 bf16* __restrict__ xemb,
                                                 bf16* __restrict__ hcat) {
  int node = blockIdx.x;
  int t = threadIdx.x;
  const float* xr = x + (size_t)node * (2 + EMB);
  xemb[(size_t)node * EMB + t] = f2bf(xr[2 + t]);
  if (t < OUTF) {
    float v0 = xr[0], v1 = xr[1];
    hcat[(size_t)node * 192 + t]        = f2bf(fmaf(v0, w1[t], b1[t]));
    hcat[(size_t)node * 192 + OUTF + t] = f2bf(fmaf(v1, w2[t], b2[t]));
  }
}

// ---------------------------------------------------------------------------
// bf16 WMMA GEMM with TDM-staged A and LDS-resident B panel.
//   C[M,NOUT] = act(A[M,K] * Bt[NOUT,K]^T + bias)
// Block = 128 threads = 4 waves; block tile = 16 rows x 64 cols.
//  - B panel (64 x K bf16, <= 48 KB) staged once into LDS.
//  - A tiles (16 x 32 bf16) DMA'd global->LDS by the Tensor Data Mover,
//    double buffered: tile i+1 in flight while tile i feeds the WMMA.
// Requires: M%16==0, K%32==0, NOUT%64==0.
// ---------------------------------------------------------------------------
template <int K, int RELU, typename OutT>
__global__ __launch_bounds__(128) void gemm_wmma_tdm(const bf16* __restrict__ A,
                                                     const bf16* __restrict__ Bt,
                                                     const float* __restrict__ bias,
                                                     OutT* __restrict__ C,
                                                     int M, int NOUT) {
  __shared__ bf16 ldsA[2][16][32];   // 2 KB double buffer (TDM destination)
  __shared__ bf16 ldsB[64][K];       // up to 48 KB weight panel
  const int tid     = threadIdx.x;
  const int wave    = tid >> 5;
  const int lane    = tid & 31;
  const int row0    = blockIdx.x * 16;
  const int colbase = blockIdx.y * 64;
  const int m       = lane & 15;
  const int half    = lane >> 4;
  const int nloc    = wave * 16 + (lane & 15);
  const int n       = colbase + nloc;

  // ---- stage B panel: 64 rows of Bt (one per output col), 16B chunks ----
  for (int idx = tid; idx < 64 * K / 8; idx += 128) {
    int r = idx / (K / 8);
    int c = (idx % (K / 8)) * 8;
    *(bf16x8*)&ldsB[r][c] = *(const bf16x8*)(Bt + (size_t)(colbase + r) * K + c);
  }

  // ---- TDM prologue: A tile 0 -> buffer 0 ----
  const bf16* arow = A + (size_t)row0 * K;
  if (tid < 32) tdm_load_a(arow, lds_off(&ldsA[0][0][0]), K, M, 16);

  v8f acc = {};
  const int nk = K / 32;
  for (int i = 0; i < nk; ++i) {
    if (tid < 32) __builtin_amdgcn_s_wait_tensorcnt(0);  // tile i landed
    __syncthreads();                                     // (also covers B, i==0)
    if (i + 1 < nk && tid < 32)                          // prefetch tile i+1
      tdm_load_a(arow + (i + 1) * 32, lds_off(&ldsA[(i + 1) & 1][0][0]), K, M, 16);

    // A fragment (ISA 16-bit 16x32 layout): elems 0..7 -> K=half*8+i,
    // elems 8..15 -> K=16+half*8+i  (two contiguous 16B LDS reads)
    bf16x8 alo = *(const bf16x8*)&ldsA[i & 1][m][half * 8];
    bf16x8 ahi = *(const bf16x8*)&ldsA[i & 1][m][16 + half * 8];
    v16bf a = __builtin_shufflevector(alo, ahi, 0, 1, 2, 3, 4, 5, 6, 7,
                                      8, 9, 10, 11, 12, 13, 14, 15);
    // B fragment from LDS panel, mirrored layout
    bf16x8 blo = *(const bf16x8*)&ldsB[nloc][i * 32 + half * 8];
    bf16x8 bhi = *(const bf16x8*)&ldsB[nloc][i * 32 + 16 + half * 8];
    v16bf b = __builtin_shufflevector(blo, bhi, 0, 1, 2, 3, 4, 5, 6, 7,
                                      8, 9, 10, 11, 12, 13, 14, 15);

    acc = __builtin_amdgcn_wmma_f32_16x16x32_bf16(
        /*neg_a=*/false, a, /*neg_b=*/false, b,
        /*c_mod=*/(short)0, acc, /*reuse_a=*/false, /*reuse_b=*/false);
    __syncthreads();   // all waves done with ldsA[i&1] before it is re-DMA'd
  }

  // epilogue: C/D layout -> row = r + 8*half, col = lane&15
  float bv = bias ? bias[n] : 0.0f;
#pragma unroll
  for (int r = 0; r < 8; ++r) {
    int row = row0 + r + half * 8;
    float v = acc[r] + bv;
    if (RELU) v = v > 0.0f ? v : 0.0f;
    store_out(&C[(size_t)row * NOUT + n], v);
  }
}

// ---------------------------------------------------------------------------
// alpha_s / alpha_d: one wave32 per node; lane j handles h[j], h[j+32]
// ---------------------------------------------------------------------------
__global__ __launch_bounds__(256) void alpha_kernel(const float* __restrict__ h,
                                                    const float* __restrict__ asrc,
                                                    const float* __restrict__ adst,
                                                    float* __restrict__ als,
                                                    float* __restrict__ ald,
                                                    int n) {
  int lane = threadIdx.x & 31;
  int node = (blockIdx.x * blockDim.x + threadIdx.x) >> 5;
  if (node >= n) return;
  const float* hr = h + (size_t)node * OUTF;
  float h0 = hr[lane], h1 = hr[lane + 32];
  float s = h0 * asrc[lane] + h1 * asrc[lane + 32];
  float d = h0 * adst[lane] + h1 * adst[lane + 32];
#pragma unroll
  for (int off = 16; off > 0; off >>= 1) {
    s += __shfl_xor(s, off, 32);
    d += __shfl_xor(d, off, 32);
  }
  if (lane == 0) { als[node] = s; ald[node] = d; }
}

// edges 0..E-1 from edge_index, edges E..E+N-1 are self loops (src=dst=e-E)
__global__ __launch_bounds__(256) void edge_score_max(const int* __restrict__ ei,
                                                      const float* __restrict__ als,
                                                      const float* __restrict__ ald,
                                                      float* __restrict__ esc,
                                                      unsigned* __restrict__ maxkey,
                                                      int ne, int ntot) {
  int e = blockIdx.x * blockDim.x + threadIdx.x;
  if (e >= ntot) return;
  int s = (e < ne) ? ei[e] : (e - ne);
  int d = (e < ne) ? ei[ne + e] : (e - ne);
  float v = als[s] + ald[d];
  v = (v > 0.0f) ? v : NEG_SLOPE * v;   // leaky relu
  esc[e] = v;
  atomicMax(&maxkey[d], f2key(v));
}

__global__ __launch_bounds__(256) void edge_exp_sum(const int* __restrict__ ei,
                                                    const unsigned* __restrict__ maxkey,
                                                    float* __restrict__ esc,
                                                    float* __restrict__ denom,
                                                    int ne, int ntot) {
  int e = blockIdx.x * blockDim.x + threadIdx.x;
  if (e >= ntot) return;
  int d = (e < ne) ? ei[ne + e] : (e - ne);
  float ex = __expf(esc[e] - key2f(maxkey[d]));
  esc[e] = ex;                          // reuse buffer: now holds exp value
  atomicAdd(&denom[d], ex);
}

// 64 lanes per edge: x3[dst, :] += (ex/denom[dst]) * h[src, :]
__global__ __launch_bounds__(256) void edge_aggregate(const int* __restrict__ ei,
                                                      const float* __restrict__ esc,
                                                      const float* __restrict__ denom,
                                                      const float* __restrict__ h,
                                                      float* __restrict__ x3,
                                                      int ne, int ntot) {
  int t = blockIdx.x * blockDim.x + threadIdx.x;
  int e = t >> 6, j = t & 63;
  if (e >= ntot) return;
  int s = (e < ne) ? ei[e] : (e - ne);
  int d = (e < ne) ? ei[ne + e] : (e - ne);
  float w = esc[e] / (denom[d] + 1e-16f);
  atomicAdd(&x3[(size_t)d * OUTF + j], w * h[(size_t)s * OUTF + j]);
}

// hcat[:,128:192] = bf16(x3 + gat_b)
__global__ __launch_bounds__(256) void concat_x3(const float* __restrict__ x3,
                                                 const float* __restrict__ gb,
                                                 bf16* __restrict__ hcat, int n) {
  int t = blockIdx.x * blockDim.x + threadIdx.x;
  if (t >= n * OUTF) return;
  int node = t >> 6, j = t & 63;
  hcat[(size_t)node * 192 + 128 + j] = f2bf(x3[t] + gb[j]);
}

// ---------------------------------------------------------------------------
// Host launcher
// ---------------------------------------------------------------------------
extern "C" void kernel_launch(void* const* d_in, const int* in_sizes, int n_in,
                              void* d_out, int out_size, void* d_ws, size_t ws_size,
                              hipStream_t stream) {
  (void)in_sizes; (void)n_in; (void)out_size; (void)ws_size;
  const int N  = N_NODES;
  const int E  = N_EDGES;
  const int EN = E + N;

  const float* x     = (const float*)d_in[0];
  const int*   ei    = (const int*)d_in[1];
  const float* w1    = (const float*)d_in[2];
  const float* b1    = (const float*)d_in[3];
  const float* w2    = (const float*)d_in[4];
  const float* b2    = (const float*)d_in[5];
  const float* gat_w = (const float*)d_in[6];
  const float* asrc  = (const float*)d_in[7];
  const float* adst  = (const float*)d_in[8];
  const float* gat_b = (const float*)d_in[9];
  const float* fc1_w = (const float*)d_in[10];
  const float* fc1_b = (const float*)d_in[11];
  const float* fc2_w = (const float*)d_in[12];
  const float* fc2_b = (const float*)d_in[13];
  const float* fc3_w = (const float*)d_in[14];
  const float* fc3_b = (const float*)d_in[15];

  char* ws = (char*)d_ws;
  size_t off = 0;
  auto carve = [&](size_t bytes) -> void* {
    void* p = ws + off;
    off = (off + bytes + 255) & ~(size_t)255;
    return p;
  };
  bf16*     xemb   = (bf16*)    carve((size_t)N * EMB * 2);     // [N,128] bf16
  bf16*     hcat   = (bf16*)    carve((size_t)N * 192 * 2);     // [N,192] bf16
  bf16*     t1     = (bf16*)    carve((size_t)N * 384 * 2);     // [N,384] bf16
  bf16*     t2     = (bf16*)    carve((size_t)N * 128 * 2);     // [N,128] bf16
  float*    h      = (float*)   carve((size_t)N * OUTF * 4);    // [N,64] f32
  float*    x3     = (float*)   carve((size_t)N * OUTF * 4);    // [N,64] f32
  float*    als    = (float*)   carve((size_t)N * 4);
  float*    ald    = (float*)   carve((size_t)N * 4);
  float*    denom  = (float*)   carve((size_t)N * 4);
  unsigned* maxkey = (unsigned*)carve((size_t)N * 4);
  float*    esc    = (float*)   carve((size_t)EN * 4);          // scores -> exp
  bf16*     gat_wt = (bf16*)    carve((size_t)64  * 128 * 2);   // [64,128]
  bf16*     fc1_wt = (bf16*)    carve((size_t)384 * 192 * 2);   // [384,192]
  bf16*     fc2_wt = (bf16*)    carve((size_t)128 * 384 * 2);   // [128,384]
  bf16*     fc3_wt = (bf16*)    carve((size_t)64  * 128 * 2);   // [64,128]

  // 1) init accumulators / softmax state
  gat_init<<<(N * OUTF + 255) / 256, 256, 0, stream>>>(x3, denom, maxkey, N);

  // 2) weight convert + transpose to bf16 [NOUT,K]
  wt_convert<<<(128 * 64  + 255) / 256, 256, 0, stream>>>(gat_w, gat_wt, 128, 64);
  wt_convert<<<(192 * 384 + 255) / 256, 256, 0, stream>>>(fc1_w, fc1_wt, 192, 384);
  wt_convert<<<(384 * 128 + 255) / 256, 256, 0, stream>>>(fc2_w, fc2_wt, 384, 128);
  wt_convert<<<(128 * 64  + 255) / 256, 256, 0, stream>>>(fc3_w, fc3_wt, 128, 64);

  // 3) node prep: x1|x2 into hcat, embedding -> bf16
  node_prep<<<N, 128, 0, stream>>>(x, w1, b1, w2, b2, xemb, hcat);

  // 4) h = xemb @ gat_w   (WMMA bf16 + TDM, f32 out, no bias/act)
  gemm_wmma_tdm<128, 0, float><<<dim3(N / 16, 1), 128, 0, stream>>>(
      xemb, gat_wt, nullptr, h, N, 64);

  // 5) attention logits per node
  alpha_kernel<<<(N * 32 + 255) / 256, 256, 0, stream>>>(h, asrc, adst, als, ald, N);

  // 6) segment softmax over incoming edges + weighted aggregation
  edge_score_max<<<(EN + 255) / 256, 256, 0, stream>>>(ei, als, ald, esc, maxkey, E, EN);
  edge_exp_sum  <<<(EN + 255) / 256, 256, 0, stream>>>(ei, maxkey, esc, denom, E, EN);
  edge_aggregate<<<(EN * 64 + 255) / 256, 256, 0, stream>>>(ei, esc, denom, h, x3, E, EN);

  // 7) hcat[:,128:192] = x3 + gat_b
  concat_x3<<<(N * OUTF + 255) / 256, 256, 0, stream>>>(x3, gat_b, hcat, N);

  // 8) MLP: fc1(relu) -> fc2(relu) -> fc3
  gemm_wmma_tdm<192, 1, bf16><<<dim3(N / 16, 384 / 64), 128, 0, stream>>>(
      hcat, fc1_wt, fc1_b, t1, N, 384);
  gemm_wmma_tdm<384, 1, bf16><<<dim3(N / 16, 128 / 64), 128, 0, stream>>>(
      t1, fc2_wt, fc2_b, t2, N, 128);
  gemm_wmma_tdm<128, 0, float><<<dim3(N / 16, 1), 128, 0, stream>>>(
      t2, fc3_wt, fc3_b, (float*)d_out, N, 64);
}
